// LPFormer_71313636983132
// MI455X (gfx1250) — matrix operand order repaired and checked by reading, not compiled, not run on hardware
//
#include <hip/hip_runtime.h>

// ---------------- problem constants ----------------
constexpr int B_ = 128, N_ = 2048, F_ = 256, HID_ = 256, H_ = 4, D_ = 64;
constexpr int SEG = 4;                 // N split into 4 segments per batch
constexpr int TILE = 128;              // nodes per LDS tile (137KB -> 2 blocks/WGP)
constexpr int TPB = 128;               // one node per thread in score phase
constexpr int NSEG = N_ / SEG;         // 512
constexpr int TILES_PER_SEG = NSEG / TILE;  // 4
constexpr int LDSTRIDE = 260;          // 16B-aligned rows (async b128) + bank skew

// ---------------- workspace layout (floats) ----------------
constexpr size_t OFF_VEC  = 0;                         // vA(1024) vB(1024) u(1024)
constexpr size_t OFF_U    = 2048;
constexpr size_t OFF_BASE = 3072;                      // B*H = 512
constexpr size_t OFF_MP   = 3584;                      // B*SEG*H = 2048 (running max)
constexpr size_t OFF_SP   = 5632;                      // B*SEG*H = 2048 (running sum)
constexpr size_t OFF_PP   = 7680;                      // B*SEG*H*F = 524288 (pooled partials)
constexpr size_t OFF_POOL = 531968;                    // B*H*F = 131072
constexpr size_t OFF_CTX  = 663040;                    // B*HID = 32768
// total = 695808 floats = ~2.8 MB

constexpr size_t LDS_BYTES =
    (size_t)(TILE * LDSTRIDE + H_ * TILE + H_ * F_ + H_ * D_ + 16 + 8) * sizeof(float);

// ---------------- CDNA5 async global->LDS (guarded; falls back cleanly) ----
#if defined(__has_builtin)
#  if __has_builtin(__builtin_amdgcn_global_load_async_to_lds_b128) && \
      __has_builtin(__builtin_amdgcn_s_wait_asynccnt)
#    define USE_ASYNC_LDS 1
#  endif
#endif
#ifndef USE_ASYNC_LDS
#  define USE_ASYNC_LDS 0
#endif
#define AS1 __attribute__((address_space(1)))
#define AS3 __attribute__((address_space(3)))
typedef int vi4 __attribute__((vector_size(16)));   // matches builtin param type

// ---------------- WMMA types ----------------
typedef __attribute__((ext_vector_type(16))) _Float16 v16h;
typedef __attribute__((ext_vector_type(8)))  float    v8f;
union U16h { v16h v; _Float16 e[16]; };
union U8f  { v8f  v; float    e[8];  };

// ============================================================
// Kernel 1a: fold attention vector into W:
//   sel 0: vA[h,f]=sum_d W[h,f,d]*aA[h,d]; sel 1: vB (aB); sel 2: u (aU)
// grid 12 blocks (sel*4+h), 256 threads (f)
// ============================================================
__global__ __launch_bounds__(256) void k_prevec(const float* __restrict__ W,
                                                const float* __restrict__ a_heads,
                                                float* __restrict__ ws) {
  const int h = blockIdx.x & 3, sel = blockIdx.x >> 2;
  const int f = threadIdx.x;
  const float* w  = W + ((size_t)(h * F_) + f) * D_;
  const float* av = a_heads + h * (4 * D_) + sel * D_;
  float s = 0.f;
#pragma unroll 4
  for (int d = 0; d < D_; ++d) s += w[d] * av[d];
  ws[OFF_VEC + (size_t)sel * (H_ * F_) + h * F_ + f] = s;
}

// ============================================================
// Kernel 1b: base[b,h] = h_a[b,:].vA[h,:] + h_b[b,:].vB[h,:]
// ============================================================
__global__ __launch_bounds__(256) void k_base(const float* __restrict__ h_a,
                                              const float* __restrict__ h_b,
                                              float* __restrict__ ws) {
  const int idx = blockIdx.x * 256 + threadIdx.x;   // 512 = B*H
  const int b = idx >> 2, h = idx & 3;
  const float* vA = ws + OFF_VEC + h * F_;
  const float* vB = ws + OFF_VEC + H_ * F_ + h * F_;
  const float* ra = h_a + (size_t)b * F_;
  const float* rb = h_b + (size_t)b * F_;
  float s = 0.f;
#pragma unroll 4
  for (int f = 0; f < F_; ++f) s += ra[f] * vA[f] + rb[f] * vB[f];
  ws[OFF_BASE + idx] = s;
}

// ============================================================
// Kernel 2: fused score + online-softmax + pooling over 128-node tiles.
// One block per (segment, batch); 4 waves; ~137KB LDS -> 2 blocks/WGP so
// one block's async global->LDS fill overlaps the other's compute.
// nodes_features is read EXACTLY ONCE from HBM.
// ============================================================
__global__ __launch_bounds__(TPB) void k_stream(const float* __restrict__ nodes,
                                                const float* __restrict__ rpes,
                                                const unsigned char* __restrict__ mask,
                                                const float* __restrict__ a_heads,
                                                float* __restrict__ ws) {
  extern __shared__ float lds[];
  float* tile = lds;                          // TILE*LDSTRIDE
  float* sc   = tile + TILE * LDSTRIDE;       // H*TILE  (softmax weights per node)
  float* uS   = sc + H_ * TILE;               // H*F
  float* arS  = uS + H_ * F_;                 // H*D (aR)
  float* wred = arS + H_ * D_;                // 16: per-wave partials
  float* tred = wred + 16;                    // 4 used: tile max / tile sum

  const int tid = threadIdx.x;
  const int seg = blockIdx.x;
  const int b   = blockIdx.y;

  const float* u = ws + OFF_U;
  for (int i = tid; i < H_ * F_; i += TPB) uS[i] = u[i];
  for (int i = tid; i < H_ * D_; i += TPB)
    arS[i] = a_heads[(i >> 6) * (4 * D_) + 3 * D_ + (i & 63)];

  float baseR[H_];
#pragma unroll
  for (int h = 0; h < H_; ++h) baseR[h] = ws[OFF_BASE + b * H_ + h];

  float runM[H_], runS[H_], pool0[H_], pool1[H_];
#pragma unroll
  for (int h = 0; h < H_; ++h) {
    runM[h] = -3.0e38f; runS[h] = 0.f; pool0[h] = 0.f; pool1[h] = 0.f;
  }

  const int wv = tid >> 5, ln = tid & 31;     // wave32: 4 waves

  for (int t = 0; t < TILES_PER_SEG; ++t) {
    const int n0 = seg * NSEG + t * TILE;
    __syncthreads();  // protect tile/sc reuse across iterations

    // ---- stage 128x256 node tile into LDS ----
    const float* gsrc = nodes + ((size_t)b * N_ + n0) * F_;
    __builtin_prefetch(gsrc + (size_t)TILE * F_ + tid * 64, 0, 0);  // next tile -> L2
#if USE_ASYNC_LDS
    for (int i = tid; i < TILE * (F_ / 4); i += TPB) {
      const int r = i >> 6, c = (i & 63) << 2;
      vi4* gp = (vi4*)(gsrc + (size_t)r * F_ + c);      // reinterpret (generic)
      vi4* lp = (vi4*)(tile + r * LDSTRIDE + c);        // reinterpret (generic)
      __builtin_amdgcn_global_load_async_to_lds_b128(
          (AS1 vi4*)gp, (AS3 vi4*)lp, 0, 0);            // explicit addrspace casts
    }
    __builtin_amdgcn_s_wait_asynccnt(0);
#else
    for (int i = tid; i < TILE * (F_ / 4); i += TPB) {
      const int r = i >> 6, c = (i & 63) << 2;
      const float4 v = *(const float4*)(gsrc + (size_t)r * F_ + c);
      *(float4*)(tile + r * LDSTRIDE + c) = v;   // 16B-aligned (stride 260)
    }
#endif
    __syncthreads();

    // ---- scores for node n0+tid (one node per thread) ----
    float e[H_];
#pragma unroll
    for (int h = 0; h < H_; ++h) e[h] = baseR[h];
    const float* rrow = rpes + ((size_t)b * N_ + n0 + tid) * D_;
    for (int d = 0; d < D_; d += 4) {
      const float4 rv = *(const float4*)(rrow + d);
#pragma unroll
      for (int h = 0; h < H_; ++h)
        e[h] += rv.x * arS[h * D_ + d]     + rv.y * arS[h * D_ + d + 1] +
                rv.z * arS[h * D_ + d + 2] + rv.w * arS[h * D_ + d + 3];
    }
    const float* trow = tile + tid * LDSTRIDE;
    for (int f = 0; f < F_; ++f) {
      const float x = trow[f];
#pragma unroll
      for (int h = 0; h < H_; ++h) e[h] += x * uS[h * F_ + f];
    }
    const bool mk = mask[(size_t)b * N_ + n0 + tid] != 0;
#pragma unroll
    for (int h = 0; h < H_; ++h) {
      e[h] = e[h] > 0.f ? e[h] : 0.2f * e[h];   // leaky_relu(0.2)
      if (!mk) e[h] = -1.0e9f;
    }

    // ---- tile max (wave32 shuffle + LDS combine over 4 waves) ----
    float wm[H_];
#pragma unroll
    for (int h = 0; h < H_; ++h) wm[h] = e[h];
#pragma unroll
    for (int o = 16; o > 0; o >>= 1) {
#pragma unroll
      for (int h = 0; h < H_; ++h) wm[h] = fmaxf(wm[h], __shfl_xor(wm[h], o, 32));
    }
    if (ln == 0) {
#pragma unroll
      for (int h = 0; h < H_; ++h) wred[h * 4 + wv] = wm[h];
    }
    __syncthreads();
    if (tid < H_) {
      float mx = -3.0e38f;
      for (int w = 0; w < 4; ++w) mx = fmaxf(mx, wred[tid * 4 + w]);
      tred[tid] = mx;
    }
    __syncthreads();

    // ---- exp weights + tile sum ----
    float newM[H_], p[H_];
#pragma unroll
    for (int h = 0; h < H_; ++h) {
      newM[h] = fmaxf(runM[h], tred[h]);
      p[h]    = __expf(e[h] - newM[h]);
      sc[h * TILE + tid] = p[h];
    }
#pragma unroll
    for (int o = 16; o > 0; o >>= 1) {
#pragma unroll
      for (int h = 0; h < H_; ++h) p[h] += __shfl_xor(p[h], o, 32);
    }
    if (ln == 0) {
#pragma unroll
      for (int h = 0; h < H_; ++h) wred[h * 4 + wv] = p[h];
    }
    __syncthreads();
    if (tid < H_) {
      float s = 0.f;
      for (int w = 0; w < 4; ++w) s += wred[tid * 4 + w];
      tred[tid] = s;
    }
    __syncthreads();

    // ---- online softmax rescale + accumulate sums ----
#pragma unroll
    for (int h = 0; h < H_; ++h) {
      const float sf = __expf(runM[h] - newM[h]);
      pool0[h] *= sf; pool1[h] *= sf;
      runS[h] = runS[h] * sf + tred[h];
      runM[h] = newM[h];
    }

    // ---- pooled: thread owns features tid and tid+128 (column reads) ----
    for (int n = 0; n < TILE; ++n) {
      const float x0 = tile[n * LDSTRIDE + tid];
      const float x1 = tile[n * LDSTRIDE + tid + 128];
#pragma unroll
      for (int h = 0; h < H_; ++h) {
        const float w = sc[h * TILE + n];
        pool0[h] += w * x0;
        pool1[h] += w * x1;
      }
    }
  }

  // ---- emit segment partials ----
  const size_t pb = (size_t)(b * SEG + seg) * H_;
#pragma unroll
  for (int h = 0; h < H_; ++h) {
    ws[OFF_PP + (pb + h) * F_ + tid]       = pool0[h];
    ws[OFF_PP + (pb + h) * F_ + tid + 128] = pool1[h];
  }
  if (tid < H_) {
    ws[OFF_MP + pb + tid] = runM[tid];
    ws[OFF_SP + pb + tid] = runS[tid];
  }
}

// ============================================================
// Kernel 2b: merge SEG segment-partials with log-sum-exp; normalize.
// ============================================================
__global__ __launch_bounds__(256) void k_merge(float* __restrict__ ws) {
  const int b = blockIdx.x, f = threadIdx.x;
#pragma unroll
  for (int h = 0; h < H_; ++h) {
    float M = -3.0e38f;
    for (int s = 0; s < SEG; ++s)
      M = fmaxf(M, ws[OFF_MP + (size_t)(b * SEG + s) * H_ + h]);
    float S = 0.f, P = 0.f;
    for (int s = 0; s < SEG; ++s) {
      const float w = __expf(ws[OFF_MP + (size_t)(b * SEG + s) * H_ + h] - M);
      S += ws[OFF_SP + (size_t)(b * SEG + s) * H_ + h] * w;
      P += ws[OFF_PP + ((size_t)(b * SEG + s) * H_ + h) * F_ + f] * w;
    }
    ws[OFF_POOL + (size_t)(b * H_ + h) * F_ + f] = P / S;
  }
}

// ============================================================
// Kernel 3 (WMMA): context[b, h*64+d] = sum_f pooled[b,h,f]*Wh_w[f, h*64+d]
// 16x16 tile per wave; v_wmma_f32_16x16x32_f16, K-loop of 8.
// A frag (16x32 f16): lane m=L&15, half=L>>4; VGPR i: K = 16*(i/4)+8*half+2*(i&3)
// B frag (32x16 f16): lane col=L&15, half;    VGPR i: K = 16*(i/4)+8*half+2*(i&3) (same as A)
// C/D (16x16 f32):    VGPR r -> M = 8*half + r, N = L&15
// ============================================================
__global__ __launch_bounds__(32) void k_ctx(const float* __restrict__ Whw,
                                            float* __restrict__ ws) {
  const int lane = threadIdx.x;
  const int m = lane & 15, half = lane >> 4;
  const int b0 = blockIdx.x << 4;
  const int j0 = blockIdx.y << 4;
  const int h  = j0 >> 6;                          // 16-col tile lies in one head
  const float* arow = ws + OFF_POOL + (size_t)(b0 + m) * (H_ * F_) + h * F_;
  U8f c;
#pragma unroll
  for (int r = 0; r < 8; ++r) c.e[r] = 0.f;
  for (int kk = 0; kk < F_; kk += 32) {
    U16h a, bb;
#pragma unroll
    for (int i = 0; i < 8; ++i) {
      const int K = kk + ((i >> 2) << 4) + (half << 3) + ((i & 3) << 1);
      a.e[2 * i]     = (_Float16)arow[K];
      a.e[2 * i + 1] = (_Float16)arow[K + 1];
    }
#pragma unroll
    for (int i = 0; i < 8; ++i) {
      const int K = kk + ((i >> 2) << 4) + (half << 3) + ((i & 3) << 1);
      bb.e[2 * i]     = (_Float16)Whw[(size_t)K * HID_ + j0 + m];
      bb.e[2 * i + 1] = (_Float16)Whw[(size_t)(K + 1) * HID_ + j0 + m];
    }
    c.v = __builtin_amdgcn_wmma_f32_16x16x32_f16(false, a.v, false, bb.v,
                                                 (short)0, c.v, false, false);
  }
  float* out = ws + OFF_CTX;
#pragma unroll
  for (int r = 0; r < 8; ++r)
    out[(size_t)(b0 + (half << 3) + r) * HID_ + j0 + m] = c.e[r];
}

// ============================================================
// Kernel 4 (WMMA): out = context @ Wo_w + Wo_b   [128x256]x[256x256]
// ============================================================
__global__ __launch_bounds__(32) void k_out(const float* __restrict__ Wow,
                                            const float* __restrict__ Wob,
                                            const float* __restrict__ ws,
                                            float* __restrict__ out) {
  const int lane = threadIdx.x;
  const int m = lane & 15, half = lane >> 4;
  const int b0 = blockIdx.x << 4;
  const int j0 = blockIdx.y << 4;
  const float* arow = ws + OFF_CTX + (size_t)(b0 + m) * HID_;
  U8f c;
#pragma unroll
  for (int r = 0; r < 8; ++r) c.e[r] = 0.f;
  for (int kk = 0; kk < HID_; kk += 32) {
    U16h a, bb;
#pragma unroll
    for (int i = 0; i < 8; ++i) {
      const int K = kk + ((i >> 2) << 4) + (half << 3) + ((i & 3) << 1);
      a.e[2 * i]     = (_Float16)arow[K];
      a.e[2 * i + 1] = (_Float16)arow[K + 1];
    }
#pragma unroll
    for (int i = 0; i < 8; ++i) {
      const int K = kk + ((i >> 2) << 4) + (half << 3) + ((i & 3) << 1);
      bb.e[2 * i]     = (_Float16)Wow[(size_t)K * HID_ + j0 + m];
      bb.e[2 * i + 1] = (_Float16)Wow[(size_t)(K + 1) * HID_ + j0 + m];
    }
    c.v = __builtin_amdgcn_wmma_f32_16x16x32_f16(false, a.v, false, bb.v,
                                                 (short)0, c.v, false, false);
  }
#pragma unroll
  for (int r = 0; r < 8; ++r)
    out[(size_t)(b0 + (half << 3) + r) * HID_ + j0 + m] = c.e[r] + Wob[j0 + m];
}

// ============================================================
extern "C" void kernel_launch(void* const* d_in, const int* in_sizes, int n_in,
                              void* d_out, int out_size, void* d_ws, size_t ws_size,
                              hipStream_t stream) {
  (void)in_sizes; (void)n_in; (void)out_size; (void)ws_size;
  const float* h_a   = (const float*)d_in[0];
  const float* h_b   = (const float*)d_in[1];
  const float* nodes = (const float*)d_in[2];
  const float* rpes  = (const float*)d_in[3];
  const unsigned char* mask = (const unsigned char*)d_in[4];  // jnp bool -> 1 byte
  const float* W     = (const float*)d_in[5];
  const float* a_h   = (const float*)d_in[6];
  const float* Whw   = (const float*)d_in[7];
  const float* Wow   = (const float*)d_in[8];
  const float* Wob   = (const float*)d_in[9];
  float* out = (float*)d_out;
  float* ws  = (float*)d_ws;

  k_prevec<<<12, 256, 0, stream>>>(W, a_h, ws);
  k_base<<<2, 256, 0, stream>>>(h_a, h_b, ws);
  k_stream<<<dim3(SEG, B_), TPB, LDS_BYTES, stream>>>(nodes, rpes, mask, a_h, ws);
  k_merge<<<B_, 256, 0, stream>>>(ws);
  k_ctx<<<dim3(B_ / 16, HID_ / 16), 32, 0, stream>>>(Whw, ws);
  k_out<<<dim3(B_ / 16, HID_ / 16), 32, 0, stream>>>(Wow, Wob, ws, out);
}
